// DotProductAttentionLayer_61314953118274
// MI455X (gfx1250) — matrix operand order repaired and checked by reading, not compiled, not run on hardware
//
#include <hip/hip_runtime.h>
#include <stdint.h>

// ---------------------------------------------------------------------------
// DotProductAttentionLayer on gfx1250 (MI455X): bf16 WMMA GEMMs, fp32 softmax.
//   query [64,512,512], key [64,512,1024], value [64,512,1024], ratios [64]
//   out   [64,512,512]
// One 512-thread workgroup (16 wave32) = one (batch, 32-row c-slab):
//   waves 0-7 -> c-tile 0 (16 rows), waves 8-15 -> c-tile 1, sharing staged K/V.
// fp32->bf16 via v_cvt_pk_bf16_f32 (1 VALU / 2 elems), vectorized b128 staging.
// ---------------------------------------------------------------------------

typedef __attribute__((ext_vector_type(16))) __bf16 v16bf;
typedef __attribute__((ext_vector_type(8)))  float  v8f;

union Frag16 { uint32_t u[8]; v16bf v; };

__device__ __forceinline__ uint32_t cvt_pk_bf16(float lo, float hi) {
  uint32_t r;
  asm volatile("v_cvt_pk_bf16_f32 %0, %1, %2" : "=v"(r) : "v"(lo), "v"(hi));
  return r;
}

constexpr int   kD = 512, kC = 512, kW = 128, kT = 1024;
constexpr float kScale = 0.044194173824159216f;   // 512^-0.5

// LDS layout (byte offsets); rows padded to spread banks
constexpr int SQ_OFF   = 0;        // uint16 sQ  [32][520]   q slab, [m][k]
constexpr int SKV_OFF  = 33280;    // uint16 sKV [1024][36]  K/V chunk
constexpr int SS_OFF   = 107008;   // float  sS  [32][1024]  logits / exp
constexpr int SW_OFF   = 238080;   // uint16 sW  [32][1032]  bf16 weights
constexpr int SRED_OFF = 304128;   // float  sRed[512]
constexpr int SMEM_BYTES = 306176; // < 320 KB/WGP

__global__ __launch_bounds__(512)
void dpa_kernel(const float* __restrict__ Q, const float* __restrict__ K,
                const float* __restrict__ V, const float* __restrict__ R,
                float* __restrict__ O) {
  extern __shared__ char smem[];
  uint16_t* sQ   = (uint16_t*)(smem + SQ_OFF);
  uint16_t* sKV  = (uint16_t*)(smem + SKV_OFF);
  float*    sS   = (float*)   (smem + SS_OFF);
  uint16_t* sW   = (uint16_t*)(smem + SW_OFF);
  float*    sRed = (float*)   (smem + SRED_OFF);

  const int tid  = threadIdx.x;
  const int wave = tid >> 5;
  const int grp  = wave >> 3;        // which 16-row c-tile this wave owns
  const int wv8  = wave & 7;         // position within the 8-wave team
  const int lane = tid & 31;
  const int half = lane >> 4;
  const int l16  = lane & 15;

  const int n  = blockIdx.x >> 4;            // batch
  const int c0 = (blockIdx.x & 15) << 5;     // 32-row c-slab base

  const float* Qn = Q + (size_t)n * kD * kC;
  const float* Kn = K + (size_t)n * kD * kT;
  const float* Vn = V + (size_t)n * kD * kT;
  float*       On = O + (size_t)n * kD * kC;

  int vw = (int)floorf(kW * R[n] + 0.5f);
  vw = vw > kW ? kW : vw;

  // Fragment K-offsets, contiguous order k = 16*half + s (16-bit A 16x32, B 32x16); pairs (k,k+1)
  int kA[8], kB[8];
#pragma unroll
  for (int r = 0; r < 8; ++r) {
    kA[r] = 2 * r + 16 * half;
    kB[r] = 2 * r + 16 * half;
  }

  // ---- stage Q slab (32 c x 512 d) -> bf16 sQ[m32][k], packed k-pairs ----
  for (int u = tid; u < 32 * 256; u += 512) {
    int m32 = u & 31, kk = (u >> 5) << 1;
    float lo = Qn[(size_t)kk * kC + c0 + m32];
    float hi = Qn[(size_t)(kk + 1) * kC + c0 + m32];
    *(uint32_t*)&sQ[m32 * 520 + kk] = cvt_pk_bf16(lo, hi);
  }
  __syncthreads();

  v8f zero = {};
  v8f acc[8];
#pragma unroll
  for (int j = 0; j < 8; ++j) acc[j] = zero;

  const int t0 = wv8 << 7;   // 128 t-columns per wave team slot

  // ============== GEMM1: S = (Q^T K) * scale, bf16 WMMA ==============
  for (int kc = 0; kc < kD / 32; ++kc) {
    if (kc) __syncthreads();
    // stage K chunk (32 d x 1024 t) -> sKV[t][d]; pack (d,d+1) pairs
    for (int u = tid; u < 16 * 256; u += 512) {
      int tq = u & 255, dp = u >> 8;
      int tt = tq << 2;
      const float* p0 = &Kn[(size_t)(kc * 32 + 2 * dp) * kT + tt];
      float4 a0 = *(const float4*)p0;
      float4 a1 = *(const float4*)(p0 + kT);
      *(uint32_t*)&sKV[(tt + 0) * 36 + 2 * dp] = cvt_pk_bf16(a0.x, a1.x);
      *(uint32_t*)&sKV[(tt + 1) * 36 + 2 * dp] = cvt_pk_bf16(a0.y, a1.y);
      *(uint32_t*)&sKV[(tt + 2) * 36 + 2 * dp] = cvt_pk_bf16(a0.z, a1.z);
      *(uint32_t*)&sKV[(tt + 3) * 36 + 2 * dp] = cvt_pk_bf16(a0.w, a1.w);
    }
    if (kc + 1 < kD / 32)                      // gfx1250 global_prefetch_b8
      __builtin_prefetch(&Kn[(size_t)((kc + 1) * 32) * kT + (tid << 1)], 0, 1);
    __syncthreads();

    Frag16 a;
#pragma unroll
    for (int r = 0; r < 8; ++r)
      a.u[r] = *(const uint32_t*)&sQ[(grp * 16 + l16) * 520 + kc * 32 + kA[r]];

#pragma unroll
    for (int j = 0; j < 8; ++j) {
      Frag16 b;
      const int tcol = t0 + j * 16 + l16;
#pragma unroll
      for (int r = 0; r < 8; ++r)
        b.u[r] = *(const uint32_t*)&sKV[tcol * 36 + kB[r]];
      acc[j] = __builtin_amdgcn_wmma_f32_16x16x32_bf16(
          false, a.v, false, b.v, (short)0, acc[j], false, false);
    }
  }

  // ---- scaled + column-masked logits -> sS (C/D layout: M=r+8*half, N=l16)
#pragma unroll
  for (int j = 0; j < 8; ++j) {
    const int tt = t0 + j * 16 + l16;
    const bool ok = (tt & (kW - 1)) < vw;
#pragma unroll
    for (int r = 0; r < 8; ++r)
      sS[(grp * 16 + r + 8 * half) * 1024 + tt] =
          ok ? acc[j][r] * kScale : -__builtin_inff();
  }
  __syncthreads();

  // ============== softmax over t: 32 rows x 16 threads each ==============
  {
    const int row = tid >> 4, sub = tid & 15;
    float* rS = sS + row * 1024;
    float mx = -__builtin_inff();
    for (int i = sub; i < kT; i += 16) mx = fmaxf(mx, rS[i]);
    sRed[tid] = mx;
    __syncthreads();
    if (sub == 0) {
      float m2 = -__builtin_inff();
      for (int i = 0; i < 16; ++i) m2 = fmaxf(m2, sRed[row * 16 + i]);
      sRed[row * 16] = m2;
    }
    __syncthreads();
    mx = sRed[row * 16];
    float sum = 0.f;
    for (int i = sub; i < kT; i += 16) {
      float e = __expf(rS[i] - mx);   // masked -inf -> 0
      rS[i] = e;
      sum += e;
    }
    __syncthreads();
    sRed[tid] = sum;
    __syncthreads();
    if (sub == 0) {
      float s2 = 0.f;
      for (int i = 0; i < 16; ++i) s2 += sRed[row * 16 + i];
      sRed[row * 16] = s2;
    }
    __syncthreads();
    const float inv = 1.f / sRed[row * 16];
    // pack stride-2 pairs -> bf16 weights (one cvt_pk + one b32 store / pair)
    for (int i = 2 * sub; i < kT; i += 32)
      *(uint32_t*)&sW[row * 1032 + i] = cvt_pk_bf16(rS[i] * inv, rS[i + 1] * inv);
  }
  __syncthreads();

  // ============== GEMM2: G = W * V^T, bf16 WMMA ==============
  v8f acc2[4];
#pragma unroll
  for (int j = 0; j < 4; ++j) acc2[j] = zero;
  const int d0 = wv8 << 6;   // 64 d-columns per wave team slot

  for (int tc = 0; tc < kT / 32; ++tc) {
    __syncthreads();
    // stage V chunk (512 d x 32 t) -> sKV[d][t]; (t,t+1) contiguous in global
    for (int u = tid; u < 512 * 8; u += 512) {
      int tq = u & 7, dd = u >> 3;
      int tl = tq << 2;
      float4 a = *(const float4*)&Vn[(size_t)dd * kT + tc * 32 + tl];
      uint32_t w0 = cvt_pk_bf16(a.x, a.y);
      uint32_t w1 = cvt_pk_bf16(a.z, a.w);
      *(uint64_t*)&sKV[dd * 36 + tl] = ((uint64_t)w1 << 32) | w0;
    }
    if (tc + 1 < kT / 32)
      __builtin_prefetch(&Vn[(size_t)(tid & 511) * kT + (tc + 1) * 32], 0, 1);
    __syncthreads();

    Frag16 a;
#pragma unroll
    for (int r = 0; r < 8; ++r)
      a.u[r] = *(const uint32_t*)&sW[(grp * 16 + l16) * 1032 + tc * 32 + kA[r]];

#pragma unroll
    for (int j = 0; j < 4; ++j) {
      Frag16 b;
      const int dc = d0 + j * 16 + l16;
#pragma unroll
      for (int r = 0; r < 8; ++r)
        b.u[r] = *(const uint32_t*)&sKV[dc * 36 + kB[r]];
      acc2[j] = __builtin_amdgcn_wmma_f32_16x16x32_bf16(
          false, a.v, false, b.v, (short)0, acc2[j], false, false);
    }
  }

  // ---- store glimpse^T into O[n][d][c] (r-stores contiguous -> b128) ----
#pragma unroll
  for (int j = 0; j < 4; ++j) {
    const int dc = d0 + j * 16 + l16;
#pragma unroll
    for (int r = 0; r < 8; ++r)
      On[(size_t)dc * kC + c0 + grp * 16 + 8 * half + r] = acc2[j][r];
  }
}

extern "C" void kernel_launch(void* const* d_in, const int* in_sizes, int n_in,
                              void* d_out, int out_size, void* d_ws, size_t ws_size,
                              hipStream_t stream) {
  (void)in_sizes; (void)n_in; (void)out_size; (void)d_ws; (void)ws_size;
  const float* Q = (const float*)d_in[0];
  const float* K = (const float*)d_in[1];
  const float* V = (const float*)d_in[2];
  const float* R = (const float*)d_in[3];
  // d_in[4], d_in[5] are h=8, w=128 (hardcoded above)
  float* O = (float*)d_out;
  dim3 grid(64 * (kC / 32));   // 1024 blocks
  dim3 block(512);             // 16 wave32
  dpa_kernel<<<grid, block, SMEM_BYTES, stream>>>(Q, K, V, R, O);
}
